// NoSharingGraphConv_83193516523812
// MI455X (gfx1250) — matrix operand order repaired and checked by reading, not compiled, not run on hardware
//
#include <hip/hip_runtime.h>

// Problem constants
#define B_BATCH 4096
#define NPTS    17
#define FIN     256
#define FOUT    256
#define KTOT    (NPTS * FIN)      // 4352 : flattened (h, n) contraction dim

// GEMM tiling
#define BM 128
#define BN 128
#define BK 32
#define LDK (BK + 8)              // 40 bf16 = 80B row stride: conflict-spread, 16B aligned

typedef __attribute__((ext_vector_type(16))) __bf16 v16bf;
typedef __attribute__((ext_vector_type(8)))  float  v8f;
typedef __attribute__((ext_vector_type(4)))  int    v4i;

__device__ __forceinline__ unsigned short f2bf(float f) {
    unsigned int u = __builtin_bit_cast(unsigned int, f);
    u += 0x7FFFu + ((u >> 16) & 1u);          // round-to-nearest-even
    return (unsigned short)(u >> 16);
}

// gfx1250: async global -> LDS copy, 16B per lane, tracked by ASYNCcnt.
// Builtin signature (from hipcc diagnostic): (v4i AS1*, v4i AS3*, imm offset, imm cpol)
__device__ __forceinline__ void async_cp_b128(const unsigned short* g, unsigned short* l) {
    __builtin_amdgcn_global_load_async_to_lds_b128(
        (__attribute__((address_space(1))) v4i*)g,
        (__attribute__((address_space(3))) v4i*)l,
        0, 0);
}

// ---------------------------------------------------------------------------
// Kernel 1: x fp32 -> bf16, 8 elems/thread, packed 16B stores.
// ---------------------------------------------------------------------------
__global__ __launch_bounds__(256) void cvt_x(const float* __restrict__ x,
                                             unsigned short* __restrict__ xb) {
    size_t i = ((size_t)blockIdx.x * 256 + threadIdx.x) * 8;
    const float4 a = *(const float4*)(x + i);
    const float4 b = *(const float4*)(x + i + 4);
    uint4 o;
    o.x = (unsigned)f2bf(a.x) | ((unsigned)f2bf(a.y) << 16);
    o.y = (unsigned)f2bf(a.z) | ((unsigned)f2bf(a.w) << 16);
    o.z = (unsigned)f2bf(b.x) | ((unsigned)f2bf(b.y) << 16);
    o.w = (unsigned)f2bf(b.z) | ((unsigned)f2bf(b.w) << 16);
    *(uint4*)(xb + i) = o;
}

// ---------------------------------------------------------------------------
// Kernel 2: Wt[w][m][h*256+n] = adj[h][w] * W[h][w][n][m]   (bf16)
// ---------------------------------------------------------------------------
__global__ __launch_bounds__(256) void wscale_t(const float* __restrict__ W,
                                                const float* __restrict__ adj,
                                                unsigned short* __restrict__ Wt) {
    unsigned idx = blockIdx.x * 256u + threadIdx.x;
    unsigned m  = idx & 255u;
    unsigned t  = idx >> 8;
    unsigned n0 = (t & 31u) * 8u;
    t >>= 5;
    unsigned h = t % 17u;
    unsigned w = t / 17u;
    float a = adj[h * 17u + w];
    const float* src = W + ((size_t)(h * 17u + w) * 256u + n0) * 256u + m;
    unsigned v[8];
#pragma unroll
    for (int j = 0; j < 8; ++j) v[j] = f2bf(src[(size_t)j * 256u] * a);
    uint4 o;
    o.x = v[0] | (v[1] << 16);
    o.y = v[2] | (v[3] << 16);
    o.z = v[4] | (v[5] << 16);
    o.w = v[6] | (v[7] << 16);
    *(uint4*)(Wt + (size_t)(w * 256u + m) * KTOT + h * 256u + n0) = o;
}

// ---------------------------------------------------------------------------
// Kernel 3: out[:,w,:] = Xb (4096x4352) @ Wt_w^T + bias,  bf16 WMMA, f32 acc.
// 256 threads = 8 waves (wave32); block tile 128x128, K-step 32.
// Global->LDS staging via GLOBAL_LOAD_ASYNC_TO_LDS_B128 (ASYNCcnt), double
// buffered in LDS; zero VGPR staging (no spills).
// ---------------------------------------------------------------------------
__global__ __launch_bounds__(256) void gat_wmma_gemm(
        const unsigned short* __restrict__ Xb,   // [4096][4352] bf16
        const unsigned short* __restrict__ Wt,   // [17][256][4352] bf16
        const float* __restrict__ bias,          // [256]
        float* __restrict__ out)                 // [4096][17][256] f32
{
    __shared__ unsigned short As[2][BM][LDK];    // 2 x 128 x 40 bf16 = 20 KB
    __shared__ unsigned short Bs[2][BN][LDK];    // 20 KB

    const int tid  = threadIdx.x;
    const int lane = tid & 31;
    const int wv   = tid >> 5;
    const int wvM  = wv & 3;          // 0..3 -> 32-row M slice
    const int wvN  = wv >> 2;         // 0..1 -> 64-col N slice

    const int mBase = blockIdx.x * BM;
    const int nBase = blockIdx.y * BN;
    const int w     = blockIdx.z;

    const unsigned short* Ag = Xb + (size_t)mBase * KTOT;
    const unsigned short* Bg = Wt + ((size_t)w * FOUT + nBase) * KTOT;

    // per-thread async segment mapping: 2 b128 for A, 2 for B per K-slab
    const int seg0 = tid;             // + i*256, i=0..1
    // row = seg>>2 (0..127), c8 = seg&3 (0..3) -> col c8*8 within 32-wide slab

    v8f acc[2][4];
    {
        v8f z = {};
#pragma unroll
        for (int mt = 0; mt < 2; ++mt)
#pragma unroll
            for (int nt = 0; nt < 4; ++nt) acc[mt][nt] = z;
    }

    // WMMA 16-bit fragment addressing (ISA 7.12.2):
    //  A (16x32): lanes 0-15 rows M=l, K {0..7,16..23}; lanes 16-31 K {8..15,24..31}
    //  B (32x16): lanes 0-15 col N=l, K {0..7,16..23}; lanes 16-31 K {8..15,24..31}
    const int aRow0 = wvM * 32 + (lane & 15);
    const int bRow0 = wvN * 64 + (lane & 15);
    const int hiSel = (lane >> 4) & 1;

    // prologue: stage slab 0 into buffer 0
#pragma unroll
    for (int i = 0; i < 2; ++i) {
        int s = i * 256 + seg0, row = s >> 2, c8 = s & 3;
        async_cp_b128(Ag + (size_t)row * KTOT + c8 * 8, &As[0][row][c8 * 8]);
        async_cp_b128(Bg + (size_t)row * KTOT + c8 * 8, &Bs[0][row][c8 * 8]);
    }

    const int NKT = KTOT / BK;  // 136
    int buf = 0;
    for (int kt = 0; kt < NKT; ++kt) {
        if (kt + 1 < NKT) {
            const int k0 = (kt + 1) * BK;
#pragma unroll
            for (int i = 0; i < 2; ++i) {
                int s = i * 256 + seg0, row = s >> 2, c8 = s & 3;
                async_cp_b128(Ag + (size_t)row * KTOT + k0 + c8 * 8,
                              &As[buf ^ 1][row][c8 * 8]);
                async_cp_b128(Bg + (size_t)row * KTOT + k0 + c8 * 8,
                              &Bs[buf ^ 1][row][c8 * 8]);
            }
            // keep the 4 just-issued copies in flight; wait for current slab
            asm volatile("s_wait_asynccnt 0x4" ::: "memory");
        } else {
            asm volatile("s_wait_asynccnt 0x0" ::: "memory");
        }
        __syncthreads();   // slab `buf` visible to all waves

        v16bf af[2], bfr[4];
#pragma unroll
        for (int mt = 0; mt < 2; ++mt) {
            const int r = aRow0 + mt * 16;
            const int c = hiSel * 8;
            union { uint4 u[2]; v16bf v; } tmp;
            tmp.u[0] = *(const uint4*)(&As[buf][r][c]);
            tmp.u[1] = *(const uint4*)(&As[buf][r][c + 16]);
            af[mt] = tmp.v;
        }
#pragma unroll
        for (int nt = 0; nt < 4; ++nt) {
            const int r = bRow0 + nt * 16;
            const int c = hiSel * 8;
            union { uint4 u[2]; v16bf v; } tmp;
            tmp.u[0] = *(const uint4*)(&Bs[buf][r][c]);
            tmp.u[1] = *(const uint4*)(&Bs[buf][r][c + 16]);
            bfr[nt] = tmp.v;
        }
#pragma unroll
        for (int mt = 0; mt < 2; ++mt)
#pragma unroll
            for (int nt = 0; nt < 4; ++nt)
                acc[mt][nt] = __builtin_amdgcn_wmma_f32_16x16x32_bf16(
                    false, af[mt], false, bfr[nt],
                    (short)0, acc[mt][nt], false, false);

        __syncthreads();   // all waves done reading `buf` before it is re-filled
        buf ^= 1;
    }

    // Epilogue: C/D layout (VGPR r: lanes 0-15 -> M=r, lanes 16-31 -> M=r+8)
#pragma unroll
    for (int nt = 0; nt < 4; ++nt) {
        const int col = nBase + wvN * 64 + nt * 16 + (lane & 15);
        const float bv = bias[col];
#pragma unroll
        for (int mt = 0; mt < 2; ++mt) {
            const int rbase = mBase + wvM * 32 + mt * 16 + hiSel * 8;
            v8f c = acc[mt][nt];
#pragma unroll
            for (int r = 0; r < 8; ++r)
                out[(size_t)(rbase + r) * (NPTS * FOUT) + (size_t)w * FOUT + col] =
                    c[r] + bv;
        }
    }
}

// ---------------------------------------------------------------------------
extern "C" void kernel_launch(void* const* d_in, const int* in_sizes, int n_in,
                              void* d_out, int out_size, void* d_ws, size_t ws_size,
                              hipStream_t stream) {
    const float* x    = (const float*)d_in[0];  // [4096][17][256]
    const float* adj  = (const float*)d_in[1];  // [17][17]
    const float* W    = (const float*)d_in[2];  // [17][17][256][256]
    const float* bias = (const float*)d_in[3];  // [256]
    float* out = (float*)d_out;                 // [4096][17][256]

    unsigned short* Xb = (unsigned short*)d_ws;                    // 4096*4352 bf16 = 35.7 MB
    unsigned short* Wt = Xb + (size_t)B_BATCH * KTOT;              // 17*256*4352 bf16 = 37.9 MB

    cvt_x<<<(B_BATCH * KTOT) / 8 / 256, 256, 0, stream>>>(x, Xb);
    wscale_t<<<(17 * 17 * 32 * 256) / 256, 256, 0, stream>>>(W, adj, Wt);
    gat_wmma_gemm<<<dim3(B_BATCH / BM, FOUT / BN, NPTS), 256, 0, stream>>>(
        Xb, Wt, bias, out);
}
